// SparseAttention_18313740550723
// MI455X (gfx1250) — matrix operand order repaired and checked by reading, not compiled, not run on hardware
//
#include <hip/hip_runtime.h>

#define D_MODEL   768
#define NUM_HEADS 12
#define HEAD_DIM  64
#define SEQ       2048
#define BATCH     2
#define WIN_HALF  256
#define TOPK      64
#define SCALE_F   0.125f   // 1/sqrt(64)
#define KDIM      768
#define NK        (KDIM / 32)
#define WROW_US   40       // LDS row pitch in ushorts: 32 data + 8 pad (80 B)

typedef __attribute__((ext_vector_type(16))) __bf16 v16bf;
typedef __attribute__((ext_vector_type(8)))  float  v8f;
typedef __attribute__((ext_vector_type(4)))  unsigned v4u;
typedef __attribute__((ext_vector_type(8)))  int      v8i;
typedef __attribute__((ext_vector_type(4)))  int      v4i;

#if defined(__has_builtin)
#if __has_builtin(__builtin_amdgcn_tensor_load_to_lds) && __has_builtin(__builtin_amdgcn_s_wait_tensorcnt)
#define HAVE_TDM 1
#endif
#endif

union Frag { uint4 u[2]; v16bf v; };

__device__ __forceinline__ unsigned short f2bf(float x) {
    unsigned u = __float_as_uint(x);
    unsigned r = (u + 0x7FFFu + ((u >> 16) & 1u)) >> 16;  // RNE
    return (unsigned short)r;
}
__device__ __forceinline__ float bf2f(unsigned h) { return __uint_as_float(h << 16); }
__device__ __forceinline__ float wave_max(float v) {
    #pragma unroll
    for (int o = 16; o > 0; o >>= 1) v = fmaxf(v, __shfl_xor(v, o, 32));
    return v;
}
__device__ __forceinline__ float wave_sum(float v) {
    #pragma unroll
    for (int o = 16; o > 0; o >>= 1) v += __shfl_xor(v, o, 32);
    return v;
}
__device__ __forceinline__ int wave_isum(int v) {
    #pragma unroll
    for (int o = 16; o > 0; o >>= 1) v += __shfl_xor(v, o, 32);
    return v;
}

#ifdef HAVE_TDM
// TDM: async 2-D tile load Global -> LDS.  tile_w elems contiguous (data_size=2B),
// tile_h rows, row stride in elements.  LDS rows padded 16B per 64B (80B pitch).
__device__ __forceinline__ void tdm_load_2d(unsigned lds_off, const void* gaddr,
                                            unsigned tile_w, unsigned tile_h,
                                            unsigned long long stride_elems)
{
    unsigned long long ga = (unsigned long long)(uintptr_t)gaddr;
    v4u g0;
    g0[0] = 1u;                                        // count=1
    g0[1] = lds_off;                                   // lds_addr (bytes)
    g0[2] = (unsigned)(ga & 0xffffffffu);              // global_addr lo
    g0[3] = (unsigned)((ga >> 32) & 0x01ffffffu) | (2u << 30);  // addr hi | type=2
    v8i g1;
    // data_size=1(2B) | pad_enable | pad_interval=3(64B) | pad_amount=3(16B)
    g1[0] = (int)((1u << 16) | (1u << 20) | (3u << 22) | (3u << 25));
    g1[1] = (int)((tile_w & 0xffffu) << 16);           // tensor_dim0[15:0] @ [63:48]
    g1[2] = (int)(((tile_w >> 16) & 0xffffu) | ((tile_h & 0xffffu) << 16));
    g1[3] = (int)(((tile_h >> 16) & 0xffffu) | ((tile_w & 0xffffu) << 16)); // tile_dim0
    g1[4] = (int)(tile_h & 0xffffu);                   // tile_dim1 | tile_dim2=0
    g1[5] = (int)(unsigned)(stride_elems & 0xffffffffu);
    g1[6] = (int)(unsigned)((stride_elems >> 32) & 0xffffu);
    g1[7] = 0;
    v4i z4 = {0, 0, 0, 0};
    v8i z8 = {0, 0, 0, 0, 0, 0, 0, 0};
    __builtin_amdgcn_tensor_load_to_lds(g0, g1, z4, z4, z8, 0);
}
#endif

// ---------------- f32 -> bf16 conversion ----------------
__global__ void cvt_bf16(const float* __restrict__ in,
                         unsigned short* __restrict__ out, int n) {
    int i = blockIdx.x * blockDim.x + threadIdx.x;
    int stride = gridDim.x * blockDim.x;
    for (; i < n; i += stride) out[i] = f2bf(in[i]);
}

// ---------------- bf16 GEMM: C[M,768] = A[M,768] @ W[768,768]^T + bias ----------------
// W row-major [N,K] == WMMA B column-major.  Weight k-slabs staged to LDS by the TDM
// (double buffered, padded pitch), A fragments register double-buffered.
// MODE 1: bf16 head-split [b,h,s,hd]; MODE 2: f32 flat [M,N].
template <int MODE>
__global__ __launch_bounds__(256) void gemm_bf16(
    const unsigned short* __restrict__ A, const unsigned short* __restrict__ W,
    const float* __restrict__ bias, unsigned short* __restrict__ outBf,
    float* __restrict__ outF)
{
    __shared__ unsigned short wtile[2][64 * WROW_US];   // 2 x 5 KB

    int lane = threadIdx.x & 31, wid = threadIdx.x >> 5;
    int m0 = blockIdx.x * 128 + wid * 16;
    int n0 = blockIdx.y * 64;
    int r     = lane & 15;
    int kb    = (lane >> 4) * 8;
    int khalf = (lane >> 4) * 16;

    v8f zero = {0.f,0.f,0.f,0.f,0.f,0.f,0.f,0.f};
    v8f acc[4] = {zero, zero, zero, zero};

    const unsigned short* wslab = W + (size_t)n0 * KDIM;

#ifdef HAVE_TDM
    if (wid == 0) {
        tdm_load_2d((unsigned)(uintptr_t)&wtile[0][0], wslab, 32u, 64u, KDIM);
        __builtin_amdgcn_s_wait_tensorcnt(0);
    }
#else
    {   // cooperative fallback: 256 threads x one 16B chunk, padded pitch
        int row = threadIdx.x >> 2, ch = threadIdx.x & 3;
        *(uint4*)&wtile[0][row * WROW_US + ch * 8] =
            *(const uint4*)(wslab + (size_t)row * KDIM + ch * 8);
    }
#endif
    __syncthreads();

    const unsigned short* arow = A + (size_t)(m0 + r) * KDIM;
    Frag fa_c, fa_n;
    fa_c.u[0] = *(const uint4*)(arow + kb);
    fa_c.u[1] = *(const uint4*)(arow + kb + 16);

    #pragma unroll 2
    for (int kk = 0; kk < NK; ++kk) {
        int nxt = (kk + 1) & 1;
        if (kk + 1 < NK) {
#ifdef HAVE_TDM
            if (wid == 0)
                tdm_load_2d((unsigned)(uintptr_t)&wtile[nxt][0],
                            wslab + (kk + 1) * 32, 32u, 64u, KDIM);
#else
            int row = threadIdx.x >> 2, ch = threadIdx.x & 3;
            *(uint4*)&wtile[nxt][row * WROW_US + ch * 8] =
                *(const uint4*)(wslab + (size_t)row * KDIM + (kk + 1) * 32 + ch * 8);
#endif
            // prefetch next A fragment (overlaps WMMA chain below)
            fa_n.u[0] = *(const uint4*)(arow + (kk + 1) * 32 + kb);
            fa_n.u[1] = *(const uint4*)(arow + (kk + 1) * 32 + kb + 16);
        }
        const unsigned short* wt = &wtile[kk & 1][0];
        #pragma unroll
        for (int jt = 0; jt < 4; ++jt) {
            const unsigned short* wr = wt + (jt * 16 + r) * WROW_US + kb;
            Frag fb;
            fb.u[0] = *(const uint4*)(wr);
            fb.u[1] = *(const uint4*)(wr + 16);
            acc[jt] = __builtin_amdgcn_wmma_f32_16x16x32_bf16(
                false, fa_c.v, false, fb.v, (short)0, acc[jt], false, false);
        }
        if (kk + 1 < NK) {
#ifdef HAVE_TDM
            if (wid == 0) __builtin_amdgcn_s_wait_tensorcnt(0);
#endif
            __syncthreads();
            fa_c = fa_n;
        }
    }

    int rb = (lane >> 4) * 8;
    #pragma unroll
    for (int jt = 0; jt < 4; ++jt) {
        int n = n0 + jt * 16 + r;
        float bv = bias[n];
        #pragma unroll
        for (int j = 0; j < 8; ++j) {
            int m = m0 + rb + j;
            float val = acc[jt][j] + bv;
            if constexpr (MODE == 2) {
                outF[(size_t)m * D_MODEL + n] = val;
            } else {   // head-split [b,h,s,d]
                int b = m >> 11, s = m & 2047;
                int h = n >> 6,  d = n & 63;
                outBf[(((size_t)b * NUM_HEADS + h) * SEQ + s) * HEAD_DIM + d] = f2bf(val);
            }
        }
    }
}

// ---------------- attention: one WG per (b,h,16 queries) ----------------
__global__ __launch_bounds__(256) void attn_kernel(
    const unsigned short* __restrict__ Q, const unsigned short* __restrict__ Km,
    const unsigned short* __restrict__ V, unsigned short* __restrict__ Ctx)
{
    __shared__ float sc[16][SEQ];       // 128 KB: full dense score rows
    __shared__ float selv[16][TOPK];
    __shared__ int   seli[16][TOPK];

    int lane = threadIdx.x & 31, wid = threadIdx.x >> 5;
    int nqt = SEQ / 16;
    int bh  = blockIdx.x / nqt;
    int q0  = (blockIdx.x % nqt) * 16;
    int b = bh / NUM_HEADS, h = bh % NUM_HEADS;
    size_t base = (size_t)bh * SEQ;

    int rr    = lane & 15;
    int kb    = (lane >> 4) * 8;
    int khalf = (lane >> 4) * 16;
    v8f zero = {0.f,0.f,0.f,0.f,0.f,0.f,0.f,0.f};

    const unsigned short* qrow = Q + (base + q0 + rr) * HEAD_DIM;
    Frag fa0, fa1;                               // Q fragments, hd 0..31 / 32..63
    fa0.u[0] = *(const uint4*)(qrow + khalf);
    fa0.u[1] = *(const uint4*)(qrow + khalf + 8);
    fa1.u[0] = *(const uint4*)(qrow + 32 + khalf);
    fa1.u[1] = *(const uint4*)(qrow + 32 + khalf + 8);

    const unsigned short* vbase = V + base * HEAD_DIM;

    // ---- Phase 1: scores via WMMA, register double-buffered K fragments ----
    {
        int kc0 = wid * 256;
        const unsigned short* krow = Km + (base + kc0 + rr) * HEAD_DIM;
        Frag c0, c1, p0, p1;
        c0.u[0] = *(const uint4*)(krow + khalf);
        c0.u[1] = *(const uint4*)(krow + khalf + 8);
        c1.u[0] = *(const uint4*)(krow + 32 + khalf);
        c1.u[1] = *(const uint4*)(krow + 32 + khalf + 8);
        #pragma unroll
        for (int t = 0; t < 16; ++t) {
            if (t < 15) {
                const unsigned short* kr = Km + (base + kc0 + (t + 1) * 16 + rr) * HEAD_DIM;
                p0.u[0] = *(const uint4*)(kr + khalf);
                p0.u[1] = *(const uint4*)(kr + khalf + 8);
                p1.u[0] = *(const uint4*)(kr + 32 + khalf);
                p1.u[1] = *(const uint4*)(kr + 32 + khalf + 8);
            }
            v8f acc = zero;
            acc = __builtin_amdgcn_wmma_f32_16x16x32_bf16(false, fa0.v, false, c0.v,
                                                          (short)0, acc, false, false);
            acc = __builtin_amdgcn_wmma_f32_16x16x32_bf16(false, fa1.v, false, c1.v,
                                                          (short)0, acc, false, false);
            int colk = kc0 + t * 16 + rr;
            int rbase = (lane >> 4) * 8;
            #pragma unroll
            for (int j = 0; j < 8; ++j) sc[rbase + j][colk] = acc[j] * SCALE_F;
            c0 = p0; c1 = p1;
        }
    }
    // warm the V window for phase 2 while other waves finish scoring
    for (int j = q0 - WIN_HALF + (int)threadIdx.x; j <= q0 + 15; j += 256) {
        if (j >= 0) __builtin_prefetch(vbase + (size_t)j * HEAD_DIM, 0, 1);
    }
    __syncthreads();

    // ---- Phases 2+3: each wave owns 2 query rows ----
    for (int rI = 0; rI < 2; ++rI) {
        int r = wid * 2 + rI;
        int q = q0 + r;

        // Windowed causal-local softmax: keys in [q-256, q]
        int jlo = q - WIN_HALF; if (jlo < 0) jlo = 0;
        int jhi = q;
        float mx = -3.4e38f;
        for (int j = jlo + lane; j <= jhi; j += 32) mx = fmaxf(mx, sc[r][j]);
        mx = wave_max(mx);
        float se = 0.f;
        for (int j = jlo + lane; j <= jhi; j += 32) se += __expf(sc[r][j] - mx);
        se = wave_sum(se);
        float aw0 = 0.f, aw1 = 0.f;   // lane covers hd dims {2*lane, 2*lane+1}
        for (int j = jlo; j <= jhi; ++j) {
            float p = __expf(sc[r][j] - mx);
            unsigned pv = *(const unsigned*)(vbase + (size_t)j * HEAD_DIM + 2 * lane);
            aw0 += p * bf2f(pv & 0xffffu);
            aw1 += p * bf2f(pv >> 16);
        }

        // Top-64 over dense scores via threshold bisection in LDS
        float gmx = -3.4e38f, gmn = 3.4e38f;
        for (int j = lane; j < SEQ; j += 32) {
            float s = sc[r][j];
            gmx = fmaxf(gmx, s); gmn = fminf(gmn, s);
        }
        gmx = wave_max(gmx);
        gmn = -wave_max(-gmn);
        float tlo = gmn - 1.0f, thi = gmx;
        for (int it = 0; it < 28; ++it) {
            float tm = 0.5f * (tlo + thi);
            int c = 0;
            for (int j = lane; j < SEQ; j += 32) c += (sc[r][j] > tm) ? 1 : 0;
            c = wave_isum(c);
            if (c > TOPK) tlo = tm; else thi = tm;
        }
        int cnt = 0;
        for (int bk = 0; bk < SEQ; bk += 32) {
            float s = sc[r][bk + lane];
            bool p = s > thi;
            unsigned msk = (unsigned)__ballot(p);
            int pos = cnt + __popc(msk & ((1u << lane) - 1u));
            if (p && pos < TOPK) { selv[r][pos] = s; seli[r][pos] = bk + lane; }
            cnt += __popc(msk);
        }
        for (int bk = 0; bk < SEQ && cnt < TOPK; bk += 32) {
            float s = sc[r][bk + lane];
            bool p = (s <= thi) && (s > tlo);
            unsigned msk = (unsigned)__ballot(p);
            int pos = cnt + __popc(msk & ((1u << lane) - 1u));
            if (p && pos < TOPK) { selv[r][pos] = s; seli[r][pos] = bk + lane; }
            cnt += __popc(msk);
        }

        float se2 = __expf(selv[r][lane] - gmx) + __expf(selv[r][lane + 32] - gmx);
        se2 = wave_sum(se2);
        float as0 = 0.f, as1 = 0.f;
        for (int t = 0; t < TOPK; ++t) {
            float p = __expf(selv[r][t] - gmx);
            int idx = seli[r][t];
            unsigned pv = *(const unsigned*)(vbase + (size_t)idx * HEAD_DIM + 2 * lane);
            as0 += p * bf2f(pv & 0xffffu);
            as1 += p * bf2f(pv >> 16);
        }

        float o0 = 0.5f * (aw0 / se + as0 / se2);
        float o1 = 0.5f * (aw1 / se + as1 / se2);
        unsigned packed = (unsigned)f2bf(o0) | ((unsigned)f2bf(o1) << 16);
        *(unsigned*)(Ctx + ((size_t)(b * SEQ + q) * D_MODEL + h * HEAD_DIM + 2 * lane)) = packed;
    }
}

// ---------------- host side ----------------
extern "C" void kernel_launch(void* const* d_in, const int* in_sizes, int n_in,
                              void* d_out, int out_size, void* d_ws, size_t ws_size,
                              hipStream_t stream) {
    const float* q  = (const float*)d_in[0];
    const float* k  = (const float*)d_in[1];
    const float* v  = (const float*)d_in[2];
    const float* Wq = (const float*)d_in[3];
    const float* bq = (const float*)d_in[4];
    const float* Wk = (const float*)d_in[5];
    const float* bk = (const float*)d_in[6];
    const float* Wv = (const float*)d_in[7];
    const float* bv = (const float*)d_in[8];
    const float* Wo = (const float*)d_in[9];
    const float* bo = (const float*)d_in[10];

    const size_t DD = (size_t)D_MODEL * D_MODEL;
    const size_t MD = (size_t)BATCH * SEQ * D_MODEL;
    unsigned short* ws  = (unsigned short*)d_ws;
    unsigned short* wqb = ws;
    unsigned short* wkb = wqb + DD;
    unsigned short* wvb = wkb + DD;
    unsigned short* wob = wvb + DD;
    unsigned short* xq  = wob + DD;
    unsigned short* xk  = xq  + MD;
    unsigned short* xv  = xk  + MD;
    unsigned short* Qb  = xv  + MD;
    unsigned short* Kb  = Qb  + MD;
    unsigned short* Vb  = Kb  + MD;
    unsigned short* Cx  = Vb  + MD;

    const int T = 256;
    cvt_bf16<<<(int)((DD + T - 1) / T), T, 0, stream>>>(Wq, wqb, (int)DD);
    cvt_bf16<<<(int)((DD + T - 1) / T), T, 0, stream>>>(Wk, wkb, (int)DD);
    cvt_bf16<<<(int)((DD + T - 1) / T), T, 0, stream>>>(Wv, wvb, (int)DD);
    cvt_bf16<<<(int)((DD + T - 1) / T), T, 0, stream>>>(Wo, wob, (int)DD);
    cvt_bf16<<<(int)((MD + T - 1) / T), T, 0, stream>>>(q, xq, (int)MD);
    cvt_bf16<<<(int)((MD + T - 1) / T), T, 0, stream>>>(k, xk, (int)MD);
    cvt_bf16<<<(int)((MD + T - 1) / T), T, 0, stream>>>(v, xv, (int)MD);

    dim3 g((BATCH * SEQ) / 128, D_MODEL / 64);   // (32, 12)
    gemm_bf16<1><<<g, 256, 0, stream>>>(xq, wqb, bq, Qb, nullptr);
    gemm_bf16<1><<<g, 256, 0, stream>>>(xk, wkb, bk, Kb, nullptr);
    gemm_bf16<1><<<g, 256, 0, stream>>>(xv, wvb, bv, Vb, nullptr);

    attn_kernel<<<BATCH * NUM_HEADS * (SEQ / 16), 256, 0, stream>>>(Qb, Kb, Vb, Cx);

    gemm_bf16<2><<<g, 256, 0, stream>>>(Cx, wob, bo, nullptr, (float*)d_out);
}
